// HybridERTQuantumCNN_31559419691091
// MI455X (gfx1250) — matrix operand order repaired and checked by reading, not compiled, not run on hardware
//
#include <hip/hip_runtime.h>
#include <hip/hip_bf16.h>

// ---------------------------------------------------------------------------
// HybridERTQuantumCNN forward for MI455X (gfx1250, wave32, WMMA).
//
// Activations live in f16, layout [B][L][C] (channel-fastest).  With the
// reduction index ordered k = j*Cin + ci, the WMMA B operand (32x16, K x N)
// is a single 32-byte contiguous global load per lane.  Weights are
// pre-packed into the CDNA5 A-fragment layout so the A operand is also one
// 32-byte contiguous load per lane.  Accumulation in fp32 via
// v_wmma_f32_16x16x32_f16.
//
// The K-loop has a uniform interior fast path (no bounds checks, no EXEC
// manipulation -> pure global_load_b128 + v_wmma) and a checked boundary
// path used only by the edge blocks of each row.
// ---------------------------------------------------------------------------

typedef __attribute__((ext_vector_type(16))) _Float16 v16h;
typedef __attribute__((ext_vector_type(8)))  _Float16 v8h;
typedef __attribute__((ext_vector_type(8)))  float    v8f;

__device__ __forceinline__ float silu_f(float t)    { return t / (1.0f + __expf(-t)); }
__device__ __forceinline__ float sigmoid_f(float t) { return 1.0f / (1.0f + __expf(-t)); }

// ---------------------------------------------------------------------------
// BN folding:  y = conv*sc + sh   with  sc = g/sqrt(v+eps),
//                                       sh = (bias - m)*sc + beta
// ---------------------------------------------------------------------------
__global__ void bnprep_kernel(const float* __restrict__ bias,
                              const float* __restrict__ bn,
                              float* __restrict__ sc, float* __restrict__ sh, int C)
{
    int c = blockIdx.x * blockDim.x + threadIdx.x;
    if (c >= C) return;
    float g = bn[c], b = bn[C + c], m = bn[2 * C + c], v = bn[3 * C + c];
    float s = g * rsqrtf(v + 1e-5f);
    sc[c] = s;
    sh[c] = (bias[c] - m) * s + b;
}

// ---------------------------------------------------------------------------
// Pack conv weights [Cout][Cin][KS] (fp32) into WMMA A-fragment layout (f16).
// Fragment (co_tile mt, k-step kt) is 512 halfs: lane*16 + e, where
//   co = mt*16 + (lane&15), half = lane>>4,
//   k  = kt*32 + half*16 + e                     (contiguous k order, matches B)
//   k  = j*CIN + ci                              (reduction ordering)
// ---------------------------------------------------------------------------
template<int CIN, int COUT, int KS>
__global__ void pack_w_kernel(const float* __restrict__ W, _Float16* __restrict__ out)
{
    constexpr int KSTEPS = CIN * KS / 32;
    constexpr int COT    = COUT / 16;
    int t = blockIdx.x * blockDim.x + threadIdx.x;
    if (t >= COT * KSTEPS * 32) return;
    int lane = t & 31;
    int kt   = (t >> 5) % KSTEPS;
    int mt   = t / (32 * KSTEPS);
    int co   = mt * 16 + (lane & 15);
    int hh   = lane >> 4;
    v16h vals;
#pragma unroll
    for (int e = 0; e < 16; ++e) {
        int k  = kt * 32 + hh * 16 + e;
        int j  = k / CIN;
        int ci = k % CIN;
        vals[e] = (_Float16)W[(co * CIN + ci) * KS + j];
    }
    *(v16h*)(out + (size_t)t * 16) = vals;
}

__global__ void zero_kernel(float* p, int n)
{
    int i = blockIdx.x * blockDim.x + threadIdx.x;
    if (i < n) p[i] = 0.0f;
}

// ---------------------------------------------------------------------------
// Conv1: Cin=1, ks=5, pad=2, + BN + SiLU, writes h_T[B][L][64] f16.
// 256 threads: 64 positions/block, 4 threads per position (16 channels each).
// ---------------------------------------------------------------------------
__global__ __launch_bounds__(256) void conv1_kernel(
    const float* __restrict__ x, const float* __restrict__ W,
    const float* __restrict__ sc, const float* __restrict__ sh,
    _Float16* __restrict__ outT, int L)
{
    int b   = blockIdx.y;
    int tid = threadIdx.x;
    int l   = blockIdx.x * 64 + (tid >> 2);
    int cg  = (tid & 3) * 16;
    float xv[5];
#pragma unroll
    for (int j = 0; j < 5; ++j) {
        int p = l + j - 2;
        xv[j] = (p >= 0 && p < L) ? x[(size_t)b * L + p] : 0.0f;
    }
    v16h o;
#pragma unroll
    for (int c = 0; c < 16; ++c) {
        int co = cg + c;
        float a = 0.0f;
#pragma unroll
        for (int j = 0; j < 5; ++j) a += xv[j] * W[co * 5 + j];
        o[c] = (_Float16)silu_f(a * sc[co] + sh[co]);
    }
    *(v16h*)(outT + ((size_t)b * L + l) * 64 + cg) = o;
}

// ---------------------------------------------------------------------------
// K-loop body, shared by interior (CHECK=false) and boundary (CHECK=true)
// paths.  xTb points at batch row b; ap is the lane's A-fragment base.
// ---------------------------------------------------------------------------
template<int CIN, int KS, int DIL, bool CHECK>
__device__ __forceinline__ v8f conv_kloop(const _Float16* __restrict__ xTb,
                                          const _Float16* __restrict__ ap,
                                          int l, int hh, int L)
{
    constexpr int KSTEPS = CIN * KS / 32;
    constexpr int JSTEPS = CIN / 32;
    constexpr int PAD    = DIL * (KS - 1) / 2;
    v8f acc = {};
#pragma unroll
    for (int kt = 0; kt < KSTEPS; ++kt) {
        int j   = kt / JSTEPS;
        int ci0 = (kt % JSTEPS) * 32 + hh * 16;
        int ls  = l + j * DIL - PAD;
        v16h bfrag;
        if constexpr (CHECK) {
            v16h z = {};
            bfrag = z;
            if (ls >= 0 && ls < L)
                bfrag = *(const v16h*)(xTb + (size_t)ls * CIN + ci0);
        } else {
            bfrag = *(const v16h*)(xTb + (size_t)ls * CIN + ci0);
        }
        v16h afrag = *(const v16h*)(ap + kt * 512);
        acc = __builtin_amdgcn_wmma_f32_16x16x32_f16(
                  false, afrag, false, bfrag, (short)0, acc, false, false);
    }
    return acc;
}

// ---------------------------------------------------------------------------
// Generic WMMA implicit-GEMM 1-D conv.
//   EPI 0: y = silu(bn(conv)); store.
//   EPI 1: y = sigmoid(bn(conv)) * comp; store.          (attention gate)
//   EPI 2: y = silu(bn(conv) + comp); global-avg-pool accumulate; no store.
// Block = 8 waves; wave w -> co_tile (w % COT), n_tile (w / COT).
// ---------------------------------------------------------------------------
template<int CIN, int COUT, int KS, int DIL, int EPI>
__global__ __launch_bounds__(256) void conv_wmma_kernel(
    const _Float16* __restrict__ xT, const _Float16* __restrict__ Apk,
    const float* __restrict__ sc, const float* __restrict__ sh,
    _Float16* __restrict__ outT, const _Float16* __restrict__ comp,
    float* __restrict__ pool, int L)
{
    constexpr int KSTEPS = CIN * KS / 32;
    constexpr int COT    = COUT / 16;
    constexpr int NT     = 8 / COT;
    constexpr int TL     = NT * 16;
    constexpr int PAD    = DIL * (KS - 1) / 2;

    int w    = threadIdx.x >> 5;
    int lane = threadIdx.x & 31;
    int cot  = w % COT;
    int nt   = w / COT;
    int b    = blockIdx.y;
    int n    = lane & 15;
    int hh   = lane >> 4;
    int lblk = blockIdx.x * TL;                 // block's first spatial index
    int l    = lblk + nt * 16 + n;

    const _Float16* xTb = xT + (size_t)b * L * CIN;
    const _Float16* ap  = Apk + (size_t)cot * KSTEPS * 512 + lane * 16;

    v8f acc;
    if (lblk >= PAD && lblk + TL + PAD <= L) {
        // interior: uniform, unchecked loads -> pure b128 + wmma inner loop
        acc = conv_kloop<CIN, KS, DIL, false>(xTb, ap, l, hh, L);
    } else {
        acc = conv_kloop<CIN, KS, DIL, true>(xTb, ap, l, hh, L);
    }

    int cobase = cot * 16 + 8 * hh;          // lane holds co = cobase + v, v=0..7

    if constexpr (EPI == 0) {
        v8h o;
#pragma unroll
        for (int v = 0; v < 8; ++v) {
            int co = cobase + v;
            o[v] = (_Float16)silu_f(acc[v] * sc[co] + sh[co]);
        }
        *(v8h*)(outT + ((size_t)b * L + l) * COUT + cobase) = o;
    } else if constexpr (EPI == 1) {
        v8h hv = *(const v8h*)(comp + ((size_t)b * L + l) * COUT + cobase);
        v8h o;
#pragma unroll
        for (int v = 0; v < 8; ++v) {
            int co = cobase + v;
            o[v] = (_Float16)(sigmoid_f(acc[v] * sc[co] + sh[co]) * (float)hv[v]);
        }
        *(v8h*)(outT + ((size_t)b * L + l) * COUT + cobase) = o;
    } else {
        v8h iv = *(const v8h*)(comp + ((size_t)b * L + l) * COUT + cobase);
#pragma unroll
        for (int v = 0; v < 8; ++v) {
            int co = cobase + v;
            float y = silu_f(acc[v] * sc[co] + sh[co] + (float)iv[v]);
            // reduce over the 16 spatial lanes of this half-wave
            y += __shfl_xor(y, 1);
            y += __shfl_xor(y, 2);
            y += __shfl_xor(y, 4);
            y += __shfl_xor(y, 8);
            if ((lane & 15) == 0) atomicAdd(pool + b * COUT + co, y);
        }
    }
}

// ---------------------------------------------------------------------------
// Tail: pooled mean -> feature map -> 4-qubit statevector sim -> FC head.
// One thread per batch element (B=128).
// ---------------------------------------------------------------------------
__global__ void tail_kernel(const float* __restrict__ g,
                            const float* __restrict__ Wfm, const float* __restrict__ bfm,
                            const float* __restrict__ qw,
                            const float* __restrict__ Wf1, const float* __restrict__ bf1,
                            const float* __restrict__ bnf1,
                            const float* __restrict__ Wf2, const float* __restrict__ bf2,
                            const float* __restrict__ bnf2,
                            const float* __restrict__ Wf3, const float* __restrict__ bf3,
                            float* __restrict__ out, int B, int L)
{
    int b = blockIdx.x * blockDim.x + threadIdx.x;
    if (b >= B) return;

    float inv = 1.0f / (float)L;
    float qin[4];
#pragma unroll
    for (int i = 0; i < 4; ++i) {
        float a = bfm[i];
        for (int c = 0; c < 128; ++c) a += g[b * 128 + c] * inv * Wfm[i * 128 + c];
        qin[i] = a;
    }

    // 4-qubit statevector (wire 0 = MSB of the flat index)
    float sr[16], si[16];
    for (int i = 0; i < 16; ++i) { sr[i] = 0.0f; si[i] = 0.0f; }
    sr[0] = 1.0f;

    auto ry = [&](float t, int wq) {
        float c = cosf(0.5f * t), s = sinf(0.5f * t);
        int m = 1 << (3 - wq);
        for (int i0 = 0; i0 < 16; ++i0) {
            if (i0 & m) continue;
            int i1 = i0 | m;
            float r0 = sr[i0], r1 = sr[i1], m0 = si[i0], m1 = si[i1];
            sr[i0] = c * r0 - s * r1;  si[i0] = c * m0 - s * m1;
            sr[i1] = s * r0 + c * r1;  si[i1] = s * m0 + c * m1;
        }
    };
    auto rz = [&](float t, int wq) {
        float c = cosf(0.5f * t), s = sinf(0.5f * t);
        int m = 1 << (3 - wq);
        for (int i0 = 0; i0 < 16; ++i0) {
            if (i0 & m) continue;
            int i1 = i0 | m;
            float r0 = sr[i0], m0 = si[i0], r1 = sr[i1], m1 = si[i1];
            sr[i0] = c * r0 + s * m0;  si[i0] = c * m0 - s * r0;   // * exp(-i t/2)
            sr[i1] = c * r1 - s * m1;  si[i1] = c * m1 + s * r1;   // * exp(+i t/2)
        }
    };
    auto cnot = [&](int cq, int tq) {
        int cm = 1 << (3 - cq), tm = 1 << (3 - tq);
        for (int i0 = 0; i0 < 16; ++i0) {
            if ((i0 & cm) && !(i0 & tm)) {
                int i1 = i0 | tm;
                float tr = sr[i0]; sr[i0] = sr[i1]; sr[i1] = tr;
                float ti = si[i0]; si[i0] = si[i1]; si[i1] = ti;
            }
        }
    };

    for (int i = 0; i < 4; ++i) ry(qin[i], i);
    for (int i = 0; i < 4; ++i) { rz(qw[i * 2 + 0], i); ry(qw[i * 2 + 1], i); }
    for (int i = 0; i < 3; ++i) cnot(i, i + 1);
    for (int i = 0; i < 4; ++i) { rz(qw[(i + 4) * 2 + 0], i); ry(qw[(i + 4) * 2 + 1], i); }

    float q[4];
#pragma unroll
    for (int i = 0; i < 4; ++i) {
        float z = 0.0f;
        for (int idx = 0; idx < 16; ++idx) {
            float p = sr[idx] * sr[idx] + si[idx] * si[idx];
            z += ((idx >> (3 - i)) & 1) ? -p : p;
        }
        q[i] = z;
    }

    float x1[64];
    for (int c = 0; c < 64; ++c) {
        float t = bf1[c];
#pragma unroll
        for (int i = 0; i < 4; ++i) t += q[i] * Wf1[c * 4 + i];
        float gg = bnf1[c], bb = bnf1[64 + c], mm = bnf1[128 + c], vv = bnf1[192 + c];
        t = (t - mm) * (gg * rsqrtf(vv + 1e-5f)) + bb;
        x1[c] = silu_f(t);
    }
    float x2[32];
    for (int c = 0; c < 32; ++c) {
        float t = bf2[c];
        for (int i = 0; i < 64; ++i) t += x1[i] * Wf2[c * 64 + i];
        float gg = bnf2[c], bb = bnf2[32 + c], mm = bnf2[64 + c], vv = bnf2[96 + c];
        t = (t - mm) * (gg * rsqrtf(vv + 1e-5f)) + bb;
        x2[c] = silu_f(t);
    }
    float o = bf3[0];
    for (int i = 0; i < 32; ++i) o += x2[i] * Wf3[i];
    out[b] = o;
}

// ---------------------------------------------------------------------------
extern "C" void kernel_launch(void* const* d_in, const int* in_sizes, int n_in,
                              void* d_out, int out_size, void* d_ws, size_t ws_size,
                              hipStream_t stream)
{
    (void)n_in; (void)out_size; (void)ws_size;
    const float* x    = (const float*)d_in[0];
    const float* W1   = (const float*)d_in[1];
    const float* b1   = (const float*)d_in[2];
    const float* bn1  = (const float*)d_in[3];
    const float* Wa1  = (const float*)d_in[4];
    const float* ba1  = (const float*)d_in[5];
    const float* bna1 = (const float*)d_in[6];
    const float* Wa2  = (const float*)d_in[7];
    const float* ba2  = (const float*)d_in[8];
    const float* bna2 = (const float*)d_in[9];
    const float* Wd1  = (const float*)d_in[10];
    const float* bd1  = (const float*)d_in[11];
    const float* bnd1 = (const float*)d_in[12];
    const float* Wd2  = (const float*)d_in[13];
    const float* bd2  = (const float*)d_in[14];
    const float* bnd2 = (const float*)d_in[15];
    const float* Wr1  = (const float*)d_in[16];
    const float* br1  = (const float*)d_in[17];
    const float* bnr1 = (const float*)d_in[18];
    const float* Wr2  = (const float*)d_in[19];
    const float* br2  = (const float*)d_in[20];
    const float* bnr2 = (const float*)d_in[21];
    const float* Wfm  = (const float*)d_in[22];
    const float* bfm  = (const float*)d_in[23];
    const float* qw   = (const float*)d_in[24];
    const float* Wf1  = (const float*)d_in[25];
    const float* bf1  = (const float*)d_in[26];
    const float* bnf1 = (const float*)d_in[27];
    const float* Wf2  = (const float*)d_in[28];
    const float* bf2  = (const float*)d_in[29];
    const float* bnf2 = (const float*)d_in[30];
    const float* Wf3  = (const float*)d_in[31];
    const float* bf3  = (const float*)d_in[32];

    const int L = 4096;
    const int B = in_sizes[0] / L;

    // ---- workspace layout (aliased; peak ~289 MB) ------------------------
    const size_t MB = 1ull << 20;
    char* ws = (char*)d_ws;
    _Float16* hT  = (_Float16*)(ws + 0);         //  64 MB  [B][L][64]
    _Float16* a1T = (_Float16*)(ws + 64 * MB);   //  32 MB  [B][L][32]
    _Float16* h2T = (_Float16*)(ws + 96 * MB);   //  64 MB  [B][L][64]
    _Float16* d1T = (_Float16*)(ws + 160 * MB);  // 128 MB  [B][L][128]
    _Float16* d2T = (_Float16*)(ws + 0);         // 128 MB  (h/a1/h2 dead)
    _Float16* r1T = (_Float16*)(ws + 160 * MB);  // 128 MB  (d1 dead)

    size_t off = 288ull * MB;
    auto alloc = [&](size_t bytes) -> char* {
        off = (off + 255) & ~255ull;
        char* p = ws + off;
        off += bytes;
        return p;
    };
    float* g = (float*)alloc((size_t)B * 128 * 4);
    float *sc1 = (float*)alloc(64 * 4),  *sh1 = (float*)alloc(64 * 4);
    float *scA1 = (float*)alloc(32 * 4), *shA1 = (float*)alloc(32 * 4);
    float *scA2 = (float*)alloc(64 * 4), *shA2 = (float*)alloc(64 * 4);
    float *scD1 = (float*)alloc(128 * 4), *shD1 = (float*)alloc(128 * 4);
    float *scD2 = (float*)alloc(128 * 4), *shD2 = (float*)alloc(128 * 4);
    float *scR1 = (float*)alloc(128 * 4), *shR1 = (float*)alloc(128 * 4);
    float *scR2 = (float*)alloc(128 * 4), *shR2 = (float*)alloc(128 * 4);
    _Float16* pkA1 = (_Float16*)alloc(2 * 14 * 512 * 2);
    _Float16* pkA2 = (_Float16*)alloc(4 * 7 * 512 * 2);
    _Float16* pkD1 = (_Float16*)alloc(8 * 6 * 512 * 2);
    _Float16* pkD2 = (_Float16*)alloc(8 * 12 * 512 * 2);
    _Float16* pkR1 = (_Float16*)alloc(8 * 12 * 512 * 2);
    _Float16* pkR2 = (_Float16*)alloc(8 * 12 * 512 * 2);

    // ---- prep: BN folding + weight packing + pool zero -------------------
    bnprep_kernel<<<1, 128, 0, stream>>>(b1,  bn1,  sc1,  sh1,  64);
    bnprep_kernel<<<1, 128, 0, stream>>>(ba1, bna1, scA1, shA1, 32);
    bnprep_kernel<<<1, 128, 0, stream>>>(ba2, bna2, scA2, shA2, 64);
    bnprep_kernel<<<1, 128, 0, stream>>>(bd1, bnd1, scD1, shD1, 128);
    bnprep_kernel<<<1, 128, 0, stream>>>(bd2, bnd2, scD2, shD2, 128);
    bnprep_kernel<<<1, 128, 0, stream>>>(br1, bnr1, scR1, shR1, 128);
    bnprep_kernel<<<1, 128, 0, stream>>>(br2, bnr2, scR2, shR2, 128);

    pack_w_kernel<64, 32, 7><<<7, 128, 0, stream>>>(Wa1, pkA1);
    pack_w_kernel<32, 64, 7><<<7, 128, 0, stream>>>(Wa2, pkA2);
    pack_w_kernel<64, 128, 3><<<12, 128, 0, stream>>>(Wd1, pkD1);
    pack_w_kernel<128, 128, 3><<<24, 128, 0, stream>>>(Wd2, pkD2);
    pack_w_kernel<128, 128, 3><<<24, 128, 0, stream>>>(Wr1, pkR1);
    pack_w_kernel<128, 128, 3><<<24, 128, 0, stream>>>(Wr2, pkR2);

    zero_kernel<<<(B * 128 + 255) / 256, 256, 0, stream>>>(g, B * 128);

    // ---- conv pipeline ---------------------------------------------------
    conv1_kernel<<<dim3(L / 64, B), 256, 0, stream>>>(x, W1, sc1, sh1, hT, L);

    conv_wmma_kernel<64, 32, 7, 1, 0><<<dim3(L / 64, B), 256, 0, stream>>>(
        hT, pkA1, scA1, shA1, a1T, nullptr, nullptr, L);
    conv_wmma_kernel<32, 64, 7, 1, 1><<<dim3(L / 32, B), 256, 0, stream>>>(
        a1T, pkA2, scA2, shA2, h2T, hT, nullptr, L);
    conv_wmma_kernel<64, 128, 3, 2, 0><<<dim3(L / 16, B), 256, 0, stream>>>(
        h2T, pkD1, scD1, shD1, d1T, nullptr, nullptr, L);
    conv_wmma_kernel<128, 128, 3, 4, 0><<<dim3(L / 16, B), 256, 0, stream>>>(
        d1T, pkD2, scD2, shD2, d2T, nullptr, nullptr, L);
    conv_wmma_kernel<128, 128, 3, 1, 0><<<dim3(L / 16, B), 256, 0, stream>>>(
        d2T, pkR1, scR1, shR1, r1T, nullptr, nullptr, L);
    conv_wmma_kernel<128, 128, 3, 1, 2><<<dim3(L / 16, B), 256, 0, stream>>>(
        r1T, pkR2, scR2, shR2, nullptr, d2T, g, L);

    // ---- tail: pool mean -> feature map -> quantum -> FC -----------------
    tail_kernel<<<(B + 127) / 128, 128, 0, stream>>>(
        g, Wfm, bfm, qw, Wf1, bf1, bnf1, Wf2, bf2, bnf2, Wf3, bf3,
        (float*)d_out, B, L);
}
